// PathConvLayer_12635793785681
// MI455X (gfx1250) — matrix operand order repaired and checked by reading, not compiled, not run on hardware
//
#include <hip/hip_runtime.h>

typedef __attribute__((ext_vector_type(16))) __bf16 v16bf;
typedef __attribute__((ext_vector_type(8)))  float  v8f;

#define N_NODES 100000
#define KNBR    50
#define FIN     128
#define FOUT    128
#define KDIM    256          // 2*FIN
#define TM      16           // nodes (rows) per block
#define LDH     (KDIM + 8)   // padded LDS row stride in ushorts (528 B, 16B-aligned, bank-spread)

// f32 -> bf16 round-to-nearest-even
__device__ __forceinline__ unsigned short f2bf(float f) {
    unsigned int u = __float_as_uint(f);
    u += 0x7FFFu + ((u >> 16) & 1u);
    return (unsigned short)(u >> 16);
}

// Repack W[256][128] (f32 row-major) into per-lane bf16 B fragments for
// v_wmma_f32_16x16x32_bf16. Fragment layout (16-bit B, 32x16, wave32):
//   lane L holds column N = L%16; K = kt*32 + (L/16)*16 + j, j = 0..15 contiguous.
// wp index: (((kt*8 + nt)*32 + lane)*16 + j)  -> 32 contiguous bytes per lane.
__global__ void pack_weight_kernel(const float* __restrict__ W,
                                   unsigned short* __restrict__ wp) {
    int t    = blockIdx.x * blockDim.x + threadIdx.x;  // 0..32767
    int j    = t & 15;
    int lane = (t >> 4) & 31;
    int nt   = (t >> 9) & 7;
    int kt   = t >> 12;
    int k = kt * 32 + (lane >> 4) * 16 + j;
    int n = nt * 16 + (lane & 15);
    wp[t] = f2bf(W[k * FOUT + n]);
}

__global__ void __launch_bounds__(256)
sage_fused_kernel(const float* __restrict__ x,
                  const long long* __restrict__ nbr,
                  const unsigned short* __restrict__ wp,
                  const float* __restrict__ bias,
                  float* __restrict__ out) {
    __shared__ __align__(16) unsigned short h[TM][LDH];   // bf16 concat(self, agg), 16 x 256

    const int node0 = blockIdx.x * TM;
    const int t     = threadIdx.x;

    // ---------------- Phase 1a: self features -> h[r][0:128] ----------------
    {
        const int r  = t >> 4;          // 0..15
        const int c8 = (t & 15) * 8;    // 0..120, 8 features per thread
        const float4* px = (const float4*)(x + (long)(node0 + r) * FIN + c8);
        float4 a = px[0];
        float4 b = px[1];
        unsigned short* dst = &h[r][c8];
        dst[0] = f2bf(a.x); dst[1] = f2bf(a.y); dst[2] = f2bf(a.z); dst[3] = f2bf(a.w);
        dst[4] = f2bf(b.x); dst[5] = f2bf(b.y); dst[6] = f2bf(b.z); dst[7] = f2bf(b.w);
    }

    // ------------- Phase 1b: mean over 50 neighbors -> h[r][128:256] -------------
    // f32 accumulation; gather rows live in L2 (x = 51 MB < 192 MB L2).
    {
        const int r  = t >> 4;
        const int c8 = (t & 15) * 8;
        const long long* nb = nbr + (long)(node0 + r) * KNBR;
        float4 s0 = make_float4(0.f, 0.f, 0.f, 0.f);
        float4 s1 = make_float4(0.f, 0.f, 0.f, 0.f);
        long long jn = nb[0];
        for (int k = 0; k < KNBR; ++k) {
            const long long j = jn;
            if (k + 1 < KNBR) {
                jn = nb[k + 1];
                __builtin_prefetch(x + (long)jn * FIN + c8, 0, 0);  // global_prefetch_b8
            }
            const float4* px = (const float4*)(x + (long)j * FIN + c8);
            float4 a = px[0];
            float4 b = px[1];
            s0.x += a.x; s0.y += a.y; s0.z += a.z; s0.w += a.w;
            s1.x += b.x; s1.y += b.y; s1.z += b.z; s1.w += b.w;
        }
        const float inv = 1.0f / (float)KNBR;
        unsigned short* dst = &h[r][FIN + c8];
        dst[0] = f2bf(s0.x * inv); dst[1] = f2bf(s0.y * inv);
        dst[2] = f2bf(s0.z * inv); dst[3] = f2bf(s0.w * inv);
        dst[4] = f2bf(s1.x * inv); dst[5] = f2bf(s1.y * inv);
        dst[6] = f2bf(s1.z * inv); dst[7] = f2bf(s1.w * inv);
    }

    __syncthreads();

    // ---------------- Phase 2: WMMA GEMM, out = relu(h @ W + b) ----------------
    // Wave w computes the 16x16 output tile at columns [16w, 16w+16).
    const int wave = t >> 5;          // 0..7 -> column tile
    const int lane = t & 31;
    const int half = lane >> 4;       // lane group (0: lanes 0-15, 1: lanes 16-31)
    const int row  = lane & 15;       // A-matrix row / C-matrix column index

    v8f acc = {};
    #pragma unroll
    for (int kt = 0; kt < 8; ++kt) {
        // A fragment (16x32 bf16, MxK): contiguous K to match packed B, lane holds
        //   elems 0..7  : K = kt*32 + half*16 +     j
        //   elems 8..15 : K = kt*32 + half*16 + 8 + j
        v16bf a;
        const int kb0 = kt * 32 + half * 16;
        const int kb1 = kt * 32 + half * 16 + 8;
        ((uint4*)&a)[0] = *(const uint4*)&h[row][kb0];
        ((uint4*)&a)[1] = *(const uint4*)&h[row][kb1];

        // B fragment (32x16 bf16): pre-packed contiguous 32 B per lane.
        v16bf b = *(const v16bf*)(wp + (((kt * 8 + wave) * 32 + lane) * 16));

        acc = __builtin_amdgcn_wmma_f32_16x16x32_bf16(
                  /*neg_a=*/false, a, /*neg_b=*/false, b,
                  /*c_mod=*/(short)0, acc, /*reuse_a=*/false, /*reuse_b=*/false);
    }

    // C/D layout: VGPR e -> M = e + half*8, N = row. Add bias, ReLU, store f32.
    const int   col = wave * 16 + row;
    const float bv  = bias[col];
    #pragma unroll
    for (int e = 0; e < 8; ++e) {
        const int m = e + half * 8;
        float v = acc[e] + bv;
        v = v > 0.0f ? v : 0.0f;
        out[(long)(node0 + m) * FOUT + col] = v;
    }
}

extern "C" void kernel_launch(void* const* d_in, const int* in_sizes, int n_in,
                              void* d_out, int out_size, void* d_ws, size_t ws_size,
                              hipStream_t stream) {
    const float*      x    = (const float*)d_in[0];
    const long long*  nbr  = (const long long*)d_in[1];
    const float*      W    = (const float*)d_in[2];
    const float*      bias = (const float*)d_in[3];
    float*            out  = (float*)d_out;
    unsigned short*   wp   = (unsigned short*)d_ws;   // 64 KB packed bf16 weights

    // Repack weights into WMMA B-fragment layout (8*8*32*16 = 32768 bf16).
    pack_weight_kernel<<<128, 256, 0, stream>>>(W, wp);

    // Fused gather-mean + concat + GEMM + bias + ReLU. 100000/16 = 6250 blocks.
    sage_fused_kernel<<<N_NODES / TM, 256, 0, stream>>>(x, nbr, wp, bias, out);
}
